// GGRU_68599217652305
// MI455X (gfx1250) — compile-verified, matrix operand order checked by reading, not run on hardware
//
#include <hip/hip_runtime.h>
#include <hip/hip_bf16.h>
#include <stdint.h>

// ---------------------------------------------------------------------------
// GGRU on MI455X (gfx1250, wave32, WMMA).
// B=64, T=512, I=256, Hd=512.
// - Input-side GEMMs hoisted out of the recurrence (batched over B*T).
// - 9 recurrent 512x512 weights packed bf16 (4.5MB, L2-resident).
// - Recurrence: ONE persistent kernel, 3 grid-barriered WMMA stages/step.
// - Per-stage A tiles staged in LDS (shared by 8 waves/block, padded rows
//   for conflict-free b128 reads); B streams from L2 with a distance-2
//   software pipeline + dual accumulators.
// ---------------------------------------------------------------------------

#define B_  64
#define T_  512
#define I_  256
#define HD_ 512

#define LDSK 520  // padded LDS row stride (u16): lane m -> bank 4m, no conflicts

typedef __attribute__((ext_vector_type(16))) __bf16 v16bf;
typedef __attribute__((ext_vector_type(8)))  __bf16 v8bf;
typedef __attribute__((ext_vector_type(8)))  float  v8f;
typedef unsigned short u16;

// Packed-weight element offsets (bf16 elements) inside workspace PK region.
// B-operand pack layout per matrix: ((ntile*kblks + kblk)*32 + lane)*16 + h
#define PKE_HZT 0
#define PKE_HRT 262144
#define PKE_HZM 524288
#define PKE_HRM 786432
#define PKE_HH  1048576
#define PKE_FZT 1310720
#define PKE_FZM 1572864
#define PKE_FRT 1835008
#define PKE_FRM 2097152
#define PKE_XZT 2359296
#define PKE_XRT 2490368
#define PKE_XZM 2621440
#define PKE_XRM 2752512
#define PKE_XH  2883584
#define PKE_FXT 3014656
#define PKE_FXM 3080192
#define PKE_END 3145728

struct Ptr9 { const float* p[9]; };

// ---------------------------- device helpers ------------------------------

__device__ __forceinline__ v8f wmma_bf16(v16bf a, v16bf b, v8f c) {
  return __builtin_amdgcn_wmma_f32_16x16x32_bf16(false, a, false, b, (short)0, c,
                                                 false, false);
}

__device__ __forceinline__ float bf2f(u16 h) {
  union { unsigned u; float f; } x;
  x.u = ((unsigned)h) << 16;
  return x.f;
}
__device__ __forceinline__ u16 f2bf(float f) {
  union { float f; unsigned u; } x;
  x.f = f;
  unsigned u = x.u + 0x7FFFu + ((x.u >> 16) & 1u);  // round-to-nearest-even
  return (u16)(u >> 16);
}
__device__ __forceinline__ float sigf(float x) {
  return 1.0f / (1.0f + __expf(-x));
}

// A-operand halves 0..7 = K[k0..k0+7], 8..15 = K[k0+16..k0+23].
__device__ __forceinline__ v16bf ldA(const u16* p) {
  v16bf r;
  ((v8bf*)&r)[0] = *(const v8bf*)(p);
  ((v8bf*)&r)[1] = *(const v8bf*)(p + 16);
  return r;
}
__device__ __forceinline__ v16bf ldB(const u16* __restrict__ p) {
  return *(const v16bf*)p;  // one contiguous 32B load per lane
}

// Global-A GEMM (throughput kernels): distance-2 pipeline, dual accumulators.
__device__ __forceinline__ void gemm_acc(v8f& acc, const u16* __restrict__ Abase,
                                         int ldk, const u16* __restrict__ Wmat,
                                         int kblks, int mtile, int ntile, int lane) {
  const int m = lane & 15, kg = lane >> 4;
  const u16* __restrict__ arow = Abase + ((mtile << 4) + m) * ldk + (kg << 3);
  const u16* __restrict__ wp = Wmat + ((ntile * kblks) << 9) + (lane << 4);
  v8f acc0 = {}, acc1 = {};
  v16bf a0, a1, a2, b0, b1, b2;
  a0 = ldA(arow);
  b0 = ldB(wp);
  if (kblks > 1) { a1 = ldA(arow + 32); b1 = ldB(wp + 512); }
#pragma unroll
  for (int kb = 0; kb < kblks; ++kb) {
    if (kb + 2 < kblks) {
      a2 = ldA(arow + ((kb + 2) << 5));
      b2 = ldB(wp + ((kb + 2) << 9));
    }
    if (kb & 1) acc1 = wmma_bf16(a0, b0, acc1);
    else        acc0 = wmma_bf16(a0, b0, acc0);
    a0 = a1; b0 = b1;
    a1 = a2; b1 = b2;
  }
  acc += acc0 + acc1;
}

// LDS-A GEMM (recurrent kernel): A from shared (ds_load, padded rows),
// B streamed from L2 with distance-2 prefetch.
__device__ __forceinline__ void gemm_lds(v8f& acc, const u16* sA,
                                         const u16* __restrict__ Wmat,
                                         int kblks, int ntile, int lane) {
  const int m = lane & 15, kg = lane >> 4;
  const u16* arow = sA + m * LDSK + (kg << 3);
  const u16* __restrict__ wp = Wmat + ((ntile * kblks) << 9) + (lane << 4);
  v8f acc0 = {}, acc1 = {};
  v16bf b0, b1, b2;
  b0 = ldB(wp);
  if (kblks > 1) b1 = ldB(wp + 512);
#pragma unroll
  for (int kb = 0; kb < kblks; ++kb) {
    if (kb + 2 < kblks) b2 = ldB(wp + ((kb + 2) << 9));
    v16bf a = ldA(arow + (kb << 5));  // LDS: short latency
    if (kb & 1) acc1 = wmma_bf16(a, b0, acc1);
    else        acc0 = wmma_bf16(a, b0, acc0);
    b0 = b1; b1 = b2;
  }
  acc += acc0 + acc1;
}

// Cooperative copy of a 16-row x 512-col bf16 slab into padded LDS.
__device__ __forceinline__ void stage_tile(u16* dst, const u16* __restrict__ src,
                                           int srcld) {
#pragma unroll
  for (int i = 0; i < 4; ++i) {
    int c = threadIdx.x + (i << 8);  // 1024 chunks of 8 u16 (16B)
    int row = c >> 6;
    int off = (c & 63) << 3;
    *(v8bf*)(dst + row * LDSK + off) = *(const v8bf*)(src + row * srcld + off);
  }
}

// Grid-wide sense barrier: bar[0]=count, bar[64]=generation (separate line).
__device__ __forceinline__ void gsync(unsigned* bar, unsigned nblocks) {
  __threadfence();
  __syncthreads();
  if (threadIdx.x == 0) {
    volatile unsigned* gen = bar + 64;
    unsigned g = *gen;
    if (atomicAdd(bar, 1u) == nblocks - 1u) {
      atomicExch(bar, 0u);
      __threadfence();
      atomicAdd((unsigned*)gen, 1u);
    } else {
      while (*gen == g) { __builtin_amdgcn_s_sleep(2); }
    }
  }
  __syncthreads();
}

// ------------------------------- kernels -----------------------------------

// Pack fp32 row-major W[K][N] -> bf16 WMMA B-operand layout.
__global__ void pack_w_kernel(Ptr9 srcs, u16* __restrict__ dst, int nmat, int K,
                              int N) {
  int idx = blockIdx.x * blockDim.x + threadIdx.x;
  int per = K * N;
  if (idx >= nmat * per) return;
  int mat = idx / per;
  int rem = idx - mat * per;
  int kblks = K >> 5;
  int ntile = rem / (kblks << 9);
  int r2 = rem - ntile * (kblks << 9);
  int kblk = r2 >> 9;
  int r3 = r2 & 511;
  int lane = r3 >> 4;
  int h = r3 & 15;
  int n = (ntile << 4) + (lane & 15);
  int kg = lane >> 4;
  int vg = h >> 1, sub = h & 1;
  int kin = ((vg < 4) ? 0 : 16) + ((vg & 3) << 1) + sub + (kg << 3);
  int k = (kblk << 5) + kin;
  dst[idx] = f2bf(srcs.p[mat][k * N + n]);
}

__global__ void cvt_x_kernel(const float* __restrict__ xt,
                             const float* __restrict__ xm,
                             u16* __restrict__ xtb, u16* __restrict__ xmb, int n) {
  for (int i = blockIdx.x * blockDim.x + threadIdx.x; i < n;
       i += gridDim.x * blockDim.x) {
    xtb[i] = f2bf(xt[i]);
    xmb[i] = f2bf(xm[i]);
  }
}

// XG = x*X_t + (1-x)*X_m,  x = sig(X_t@fus_W_xt + X_m@fus_W_xm + fus_xbias)
__global__ void xgate_kernel(const u16* __restrict__ xtb,
                             const u16* __restrict__ xmb,
                             const u16* __restrict__ pk,
                             const float* __restrict__ xbias,
                             u16* __restrict__ xg) {
  const int lane = threadIdx.x & 31;
  const int gw = (blockIdx.x * blockDim.x + threadIdx.x) >> 5;
  const int nw = (gridDim.x * blockDim.x) >> 5;
  const u16* wxt = pk + PKE_FXT;
  const u16* wxm = pk + PKE_FXM;
  const int JOBS = (B_ * T_ / 16) * (I_ / 16);  // 2048 * 16
  for (int job = gw; job < JOBS; job += nw) {
    int mtile = job >> 4, ntile = job & 15;
    v8f acc = {};
    gemm_acc(acc, xtb, I_, wxt, I_ / 32, mtile, ntile, lane);
    gemm_acc(acc, xmb, I_, wxm, I_ / 32, mtile, ntile, lane);
    int col = (ntile << 4) + (lane & 15);
    int kg = lane >> 4;
    float bias = xbias[col];
#pragma unroll
    for (int e = 0; e < 8; ++e) {
      int row = (mtile << 4) + e + (kg << 3);
      float x = sigf(acc[e] + bias);
      float a = bf2f(xtb[row * I_ + col]);
      float b = bf2f(xmb[row * I_ + col]);
      xg[row * I_ + col] = f2bf(x * a + (1.0f - x) * b);
    }
  }
}

// 5 hoisted projections: A_* = X @ W_x* + b_*, stored [T][B][Hd] in bf16.
__global__ void inproj_kernel(const u16* __restrict__ xtb,
                              const u16* __restrict__ xmb,
                              const u16* __restrict__ xg,
                              const u16* __restrict__ pk,
                              const float* __restrict__ bzt,
                              const float* __restrict__ brt,
                              const float* __restrict__ bzm,
                              const float* __restrict__ brm,
                              const float* __restrict__ bh,
                              u16* __restrict__ azt, u16* __restrict__ art,
                              u16* __restrict__ azm, u16* __restrict__ arm,
                              u16* __restrict__ ah) {
  const int lane = threadIdx.x & 31;
  const int gw = (blockIdx.x * blockDim.x + threadIdx.x) >> 5;
  const int nw = (gridDim.x * blockDim.x) >> 5;
  const int TPG = (B_ * T_ / 16) * (HD_ / 16);  // 2048*32 tiles per gemm
  for (int job = gw; job < 5 * TPG; job += nw) {
    int g = job / TPG;
    int r = job - g * TPG;
    int mtile = r >> 5, ntile = r & 31;
    const u16 *A, *W;
    const float* bias;
    u16* D;
    switch (g) {
      case 0:  A = xtb; W = pk + PKE_XZT; bias = bzt; D = azt; break;
      case 1:  A = xtb; W = pk + PKE_XRT; bias = brt; D = art; break;
      case 2:  A = xmb; W = pk + PKE_XZM; bias = bzm; D = azm; break;
      case 3:  A = xmb; W = pk + PKE_XRM; bias = brm; D = arm; break;
      default: A = xg;  W = pk + PKE_XH;  bias = bh;  D = ah;  break;
    }
    v8f acc = {};
    gemm_acc(acc, A, I_, W, I_ / 32, mtile, ntile, lane);
    int col = (ntile << 4) + (lane & 15);
    int kg = lane >> 4;
    float bv = bias[col];
#pragma unroll
    for (int e = 0; e < 8; ++e) {
      int row = (mtile << 4) + e + (kg << 3);  // row = b*T + t
      int bb = row >> 9, tt = row & (T_ - 1);
      D[((tt * B_) + bb) * HD_ + col] = f2bf(acc[e] + bv);
    }
  }
}

__global__ void init_state(float* __restrict__ hf, u16* __restrict__ hb,
                           unsigned* __restrict__ bar, int n) {
  int i = blockIdx.x * blockDim.x + threadIdx.x;
  if (i < n) { hf[i] = 0.0f; hb[i] = 0; }
  if (i < 128) bar[i] = 0u;
}

// Persistent recurrent kernel: 64 blocks x 256 threads = 512 wave32s.
// Block mapping gives all 8 waves of a block the SAME A slab per stage.
__global__ void __launch_bounds__(256) recur_kernel(
    const u16* __restrict__ pk,
    const u16* __restrict__ azt, const u16* __restrict__ art,
    const u16* __restrict__ azm, const u16* __restrict__ arm,
    const u16* __restrict__ ah,
    const float* __restrict__ fzbias, const float* __restrict__ frbias,
    float* __restrict__ hf, u16* __restrict__ hb,
    u16* __restrict__ ztb, u16* __restrict__ rtb,
    u16* __restrict__ zmb, u16* __restrict__ rmb,
    float* __restrict__ zf, u16* __restrict__ rhb,
    float* __restrict__ out, float* __restrict__ hlast,
    unsigned* __restrict__ bar) {
  __shared__ u16 smem[2 * 16 * LDSK];  // 33,280 B
  const int lane = threadIdx.x & 31;
  const int wv = threadIdx.x >> 5;  // 0..7
  const int blk = blockIdx.x;       // 0..63
  const unsigned NB = 64;
  const int m = lane & 15, kg = lane >> 4;

  // stage 1: blk -> (gate g, mtile), 8 waves -> 8 ntiles of a 32-wide group
  const int s1_g = blk >> 4;
  const int s1_mt = (blk >> 2) & 3;
  const int s1_nt = ((blk & 3) << 3) + wv;
  const u16* s1_W = pk + s1_g * 262144;  // hz_t,hr_t,hz_m,hr_m
  const u16* s1_pre = (s1_g == 0) ? azt : (s1_g == 1) ? art
                      : (s1_g == 2) ? azm : arm;
  u16* s1_dst = (s1_g == 0) ? ztb : (s1_g == 1) ? rtb
                : (s1_g == 2) ? zmb : rmb;
  // stage 2: blocks 0..31 -> (o, mtile)
  const int s2_o = (blk >> 4) & 1;
  const int s2_mt = (blk >> 2) & 3;
  const int s2_nt = ((blk & 3) << 3) + wv;
  // stage 3: blocks 0..15 -> mtile
  const int s3_mt = blk >> 2;
  const int s3_nt = ((blk & 3) << 3) + wv;

  for (int t = 0; t < T_; ++t) {
    // ---- stage 1: Z_t,R_t,Z_m,R_m = sig(H @ W_h* + A_pre) ----
    {
      stage_tile(smem, hb + (s1_mt << 4) * HD_, HD_);
      __syncthreads();
      v8f acc = {};
      gemm_lds(acc, smem, s1_W, HD_ / 32, s1_nt, lane);
      int col = (s1_nt << 4) + m;
      const u16* ps = s1_pre + (t * B_) * HD_;
#pragma unroll
      for (int e = 0; e < 8; ++e) {
        int row = (s1_mt << 4) + e + (kg << 3);
        float v = acc[e] + bf2f(ps[row * HD_ + col]);
        s1_dst[row * HD_ + col] = f2bf(sigf(v));
      }
    }
    gsync(bar, NB);

    // ---- stage 2: z,r fusion gates -> Z (fp32), RH=R*H (bf16) ----
    if (blk < 32) {
      const u16 *A1, *A2, *W1, *W2;
      const float* fb;
      if (s2_o == 0) { A1 = ztb; A2 = zmb; W1 = pk + PKE_FZT; W2 = pk + PKE_FZM; fb = fzbias; }
      else           { A1 = rtb; A2 = rmb; W1 = pk + PKE_FRT; W2 = pk + PKE_FRM; fb = frbias; }
      stage_tile(smem,             A1 + (s2_mt << 4) * HD_, HD_);
      stage_tile(smem + 16 * LDSK, A2 + (s2_mt << 4) * HD_, HD_);
      __syncthreads();
      v8f acc = {};
      gemm_lds(acc, smem,             W1, HD_ / 32, s2_nt, lane);
      gemm_lds(acc, smem + 16 * LDSK, W2, HD_ / 32, s2_nt, lane);
      int col = (s2_nt << 4) + m;
      float bias = fb[col];
#pragma unroll
      for (int e = 0; e < 8; ++e) {
        int rl = e + (kg << 3);             // row within the staged slab
        int row = (s2_mt << 4) + rl;
        int idx = row * HD_ + col;
        float g = sigf(acc[e] + bias);
        float va = bf2f(smem[rl * LDSK + col]);              // Z_t / R_t
        float vb = bf2f(smem[16 * LDSK + rl * LDSK + col]);  // Z_m / R_m
        float fused = g * va + (1.0f - g) * vb;
        if (s2_o == 0) zf[idx] = fused;                   // Z (blend weight)
        else           rhb[idx] = f2bf(fused * hf[idx]);  // R*H
      }
    }
    gsync(bar, NB);

    // ---- stage 3: H_tilde = tanh(RH @ W_hh + A_h); H = Z*H+(1-Z)*Ht ----
    if (blk < 16) {
      stage_tile(smem, rhb + (s3_mt << 4) * HD_, HD_);
      __syncthreads();
      v8f acc = {};
      gemm_lds(acc, smem, pk + PKE_HH, HD_ / 32, s3_nt, lane);
      int col = (s3_nt << 4) + m;
      const u16* ps = ah + (t * B_) * HD_;
#pragma unroll
      for (int e = 0; e < 8; ++e) {
        int row = (s3_mt << 4) + e + (kg << 3);  // row == batch index b
        int idx = row * HD_ + col;
        float ht = tanhf(acc[e] + bf2f(ps[idx]));
        float z = zf[idx];
        float hn = z * hf[idx] + (1.0f - z) * ht;
        hf[idx] = hn;
        hb[idx] = f2bf(hn);
        out[(row * T_ + t) * HD_ + col] = hn;
        if (t == T_ - 1) hlast[idx] = hn;
      }
    }
    gsync(bar, NB);
  }
}

// ------------------------------- launcher ----------------------------------

extern "C" void kernel_launch(void* const* d_in, const int* in_sizes, int n_in,
                              void* d_out, int out_size, void* d_ws, size_t ws_size,
                              hipStream_t stream) {
  (void)in_sizes; (void)n_in; (void)out_size; (void)ws_size;
  const float* tt = (const float*)d_in[0];
  const float* mk = (const float*)d_in[1];
  const float *W_xz_t = (const float*)d_in[2],  *W_xr_t = (const float*)d_in[3],
              *W_xz_m = (const float*)d_in[4],  *W_xr_m = (const float*)d_in[5],
              *W_xh   = (const float*)d_in[6],
              *W_hz_t = (const float*)d_in[7],  *W_hr_t = (const float*)d_in[8],
              *W_hz_m = (const float*)d_in[9],  *W_hr_m = (const float*)d_in[10],
              *W_hh   = (const float*)d_in[11],
              *b_z_t  = (const float*)d_in[12], *b_r_t  = (const float*)d_in[13],
              *b_z_m  = (const float*)d_in[14], *b_r_m  = (const float*)d_in[15],
              *b_h    = (const float*)d_in[16],
              *fW_rt  = (const float*)d_in[17], *fW_rm  = (const float*)d_in[18],
              *fW_zt  = (const float*)d_in[19], *fW_zm  = (const float*)d_in[20],
              *fW_xt  = (const float*)d_in[21], *fW_xm  = (const float*)d_in[22],
              *f_rb   = (const float*)d_in[23], *f_zb   = (const float*)d_in[24],
              *f_xb   = (const float*)d_in[25];

  // Workspace carve-up (all chunks are multiples of 256B).
  char* w = (char*)d_ws;
  u16*  pk  = (u16*)w;   w += (size_t)PKE_END * 2;        // 6.0MB
  u16*  xtb = (u16*)w;   w += (size_t)B_ * T_ * I_ * 2;   // 16MB
  u16*  xmb = (u16*)w;   w += (size_t)B_ * T_ * I_ * 2;
  u16*  xgb = (u16*)w;   w += (size_t)B_ * T_ * I_ * 2;
  u16*  azt = (u16*)w;   w += (size_t)B_ * T_ * HD_ * 2;  // 32MB each
  u16*  art = (u16*)w;   w += (size_t)B_ * T_ * HD_ * 2;
  u16*  azm = (u16*)w;   w += (size_t)B_ * T_ * HD_ * 2;
  u16*  arm = (u16*)w;   w += (size_t)B_ * T_ * HD_ * 2;
  u16*  ah  = (u16*)w;   w += (size_t)B_ * T_ * HD_ * 2;
  float* hf = (float*)w; w += (size_t)B_ * HD_ * 4;
  u16*  hb  = (u16*)w;   w += (size_t)B_ * HD_ * 2;
  u16*  ztb = (u16*)w;   w += (size_t)B_ * HD_ * 2;
  u16*  rtb = (u16*)w;   w += (size_t)B_ * HD_ * 2;
  u16*  zmb = (u16*)w;   w += (size_t)B_ * HD_ * 2;
  u16*  rmb = (u16*)w;   w += (size_t)B_ * HD_ * 2;
  float* zf = (float*)w; w += (size_t)B_ * HD_ * 4;
  u16*  rhb = (u16*)w;   w += (size_t)B_ * HD_ * 2;
  unsigned* bar = (unsigned*)w; w += 512;

  // 1) pack weights to bf16 B-operand layout
  Ptr9 hh = {}; hh.p[0]=W_hz_t; hh.p[1]=W_hr_t; hh.p[2]=W_hz_m; hh.p[3]=W_hr_m;
  hh.p[4]=W_hh; hh.p[5]=fW_zt; hh.p[6]=fW_zm; hh.p[7]=fW_rt; hh.p[8]=fW_rm;
  Ptr9 ihm = {}; ihm.p[0]=W_xz_t; ihm.p[1]=W_xr_t; ihm.p[2]=W_xz_m;
  ihm.p[3]=W_xr_m; ihm.p[4]=W_xh;
  Ptr9 iim = {}; iim.p[0]=fW_xt; iim.p[1]=fW_xm;
  pack_w_kernel<<<(9 * 262144 + 255) / 256, 256, 0, stream>>>(hh,  pk, 9, 512, 512);
  pack_w_kernel<<<(5 * 131072 + 255) / 256, 256, 0, stream>>>(ihm, pk + PKE_XZT, 5, 256, 512);
  pack_w_kernel<<<(2 * 65536 + 255) / 256, 256, 0, stream>>>(iim, pk + PKE_FXT, 2, 256, 256);

  // 2) activations to bf16
  cvt_x_kernel<<<4096, 256, 0, stream>>>(tt, mk, xtb, xmb, B_ * T_ * I_);

  // 3) hoisted x-fusion gate + input projections (batched WMMA GEMMs)
  xgate_kernel<<<2048, 256, 0, stream>>>(xtb, xmb, pk, f_xb, xgb);
  inproj_kernel<<<4096, 256, 0, stream>>>(xtb, xmb, xgb, pk, b_z_t, b_r_t, b_z_m,
                                          b_r_m, b_h, azt, art, azm, arm, ah);

  // 4) H0 = 0, barrier reset
  init_state<<<(B_ * HD_ + 255) / 256, 256, 0, stream>>>(hf, hb, bar, B_ * HD_);

  // 5) persistent recurrence: 512 waves, 3 WMMA stages + grid barriers per step
  float* out   = (float*)d_out;
  float* hlast = out + (size_t)B_ * T_ * HD_;
  recur_kernel<<<64, 256, 0, stream>>>(pk, azt, art, azm, arm, ah, f_zb, f_rb,
                                       hf, hb, ztb, rtb, zmb, rmb, zf, rhb,
                                       out, hlast, bar);
}
